// ShareNet_50302656971595
// MI455X (gfx1250) — matrix operand order repaired and checked by reading, not compiled, not run on hardware
//
#include <hip/hip_runtime.h>
#include <hip/hip_bf16.h>

// ---------------- model constants ----------------
#define BATCH     64
#define L_ITEM    6
#define L_PL      300
#define EMBED     256
#define HEADS     8
#define HD        32
#define LAYERS    6
#define OBS_DIM   2154
#define SINGLE    718
#define AREA      100
#define FACTORF   0.1f
#define INV_SCALE 0.0625f      // 1/sqrt(256)
#define LN_EPSF   1e-5f
#define KMAX      320          // max padded key length (300 -> 320)

typedef __attribute__((ext_vector_type(16))) __bf16 v16bf;
typedef __attribute__((ext_vector_type(8)))  __bf16 v8bf;
typedef __attribute__((ext_vector_type(8)))  float  v8f;

__device__ __forceinline__ __bf16 f2bf(float f) {
  unsigned u = __builtin_bit_cast(unsigned, f);
  u += 0x7FFFu + ((u >> 16) & 1u);            // round-to-nearest-even
  unsigned short h = (unsigned short)(u >> 16);
  return __builtin_bit_cast(__bf16, h);
}

#define WMMA_BF16(A, B, C) \
  __builtin_amdgcn_wmma_f32_16x16x32_bf16(false, (A), false, (B), (short)0, (C), false, false)

// A-fragment from bf16 row-major data: lane needs K offsets
// {half*16 + 0..7} and {half*16 + 8..15} -> two 16B loads, no conversion.
__device__ __forceinline__ v16bf load_abf(const __bf16* __restrict__ p, int half) {
  const v8bf lo = *(const v8bf*)(p + (half << 4));
  const v8bf hi = *(const v8bf*)(p + 8 + (half << 4));
  v16bf v;
#pragma unroll
  for (int e = 0; e < 8; ++e) { v[e] = lo[e]; v[8 + e] = hi[e]; }
  return v;
}

// ---------------------------------------------------------------------------
// Pack W[K,N] (f32, row-major) into bf16 WMMA B-fragments.
// Fragment f = (k/32)*(N/16) + (n/16); element e of lane (lr,half) is
// W[kt*32 + half*16 + e][nt*16 + lr]; each lane owns 32 contiguous bytes.
// ---------------------------------------------------------------------------
__global__ void __launch_bounds__(256)
k_pack_w(const float* __restrict__ W, __bf16* __restrict__ Wp, int K, int N)
{
  const int i = blockIdx.x * 256 + threadIdx.x;
  if (i >= K * N) return;
  const int f = i >> 9, r = i & 511;
  const int lane = r >> 4, e = r & 15;
  const int lr = lane & 15, half = lane >> 4;
  const int ntiles = N >> 4;
  const int kt = f / ntiles, nt = f - kt * ntiles;
  const int kk = kt * 32 + (half << 4) + e;
  const int nn = (nt << 4) + lr;
  Wp[i] = f2bf(W[(size_t)kk * N + nn]);
}

// ---------------------------------------------------------------------------
// GEMM: act( A[M,K]@W[K,N] + bias ) -> optional f32 Cf and/or bf16 Cb.
// A is bf16 row-major; W pre-packed bf16 fragments.
// One wave computes a 32 x (16*NT) tile. M,K mult of 32; N mult of 16*NT.
// ---------------------------------------------------------------------------
template <int NT>
__global__ void __launch_bounds__(32)
k_gemm(const __bf16* __restrict__ A, const __bf16* __restrict__ Wp,
       const float* __restrict__ bias, float* __restrict__ Cf,
       __bf16* __restrict__ Cb, int M, int N, int K, int act)
{
  const int lane = threadIdx.x;
  const int lr   = lane & 15;
  const int half = lane >> 4;
  const int m0 = blockIdx.y * 32;
  const int n0 = blockIdx.x * (16 * NT);
  const int ntiles = N >> 4;

  v8f acc0[NT], acc1[NT];
#pragma unroll
  for (int t = 0; t < NT; ++t) { acc0[t] = (v8f){}; acc1[t] = (v8f){}; }

  const __bf16* Arow0 = A + (size_t)(m0 + lr) * K;
  const __bf16* Arow1 = A + (size_t)(m0 + 16 + lr) * K;

  for (int kt = 0; kt < K; kt += 32) {
    const v16bf a0 = load_abf(Arow0 + kt, half);
    const v16bf a1 = load_abf(Arow1 + kt, half);
    const size_t fbase = (size_t)(kt >> 5) * ntiles + (n0 >> 4);
    if (kt + 32 < K) {   // prefetch next K-step (global_prefetch path)
      __builtin_prefetch(Arow0 + kt + 32, 0, 3);
      __builtin_prefetch(Arow1 + kt + 32, 0, 3);
      __builtin_prefetch(Wp + ((fbase + ntiles) << 9) + (lane << 4), 0, 3);
    }
#pragma unroll
    for (int t = 0; t < NT; ++t) {
      const v16bf b = *(const v16bf*)(Wp + ((fbase + t) << 9) + (lane << 4));
      acc0[t] = WMMA_BF16(a0, b, acc0[t]);
      acc1[t] = WMMA_BF16(a1, b, acc1[t]);
    }
  }

#pragma unroll
  for (int t = 0; t < NT; ++t) {
    const int col = n0 + t * 16 + lr;
    const float bi = bias ? bias[col] : 0.f;
#pragma unroll
    for (int r = 0; r < 8; ++r) {
      const int row0 = m0 + r + half * 8;
      float v0 = acc0[t][r] + bi;
      float v1 = acc1[t][r] + bi;
      if (act) { v0 = fmaxf(v0, 0.f); v1 = fmaxf(v1, 0.f); }
      if (Cf) { Cf[(size_t)row0 * N + col] = v0; Cf[(size_t)(row0 + 16) * N + col] = v1; }
      if (Cb) { Cb[(size_t)row0 * N + col] = f2bf(v0); Cb[(size_t)(row0 + 16) * N + col] = f2bf(v1); }
    }
  }
}

// ---------------------------------------------------------------------------
// Transpose + pad V: vh[n, key, h*32+d] -> vt[n, h, d, keypad]   (bf16)
// ---------------------------------------------------------------------------
__global__ void __launch_bounds__(256)
k_vtrans(const __bf16* __restrict__ vh, __bf16* __restrict__ vt, int Lk, int KPAD)
{
  const int idx = blockIdx.x * 256 + threadIdx.x;
  const int total = BATCH * HEADS * HD * KPAD;
  if (idx >= total) return;
  const int key = idx % KPAD;
  int rest = idx / KPAD;
  const int d = rest & 31; rest >>= 5;
  const int h = rest & 7;  const int n = rest >> 3;
  vt[idx] = (key < Lk) ? vh[(((size_t)n * Lk + key) * HEADS + h) * HD + d]
                       : __builtin_bit_cast(__bf16, (unsigned short)0);
}

// ---------------------------------------------------------------------------
// Fused attention: one wave per (batch n, head h, 16-query tile).
// qh/kh bf16 [n, token, h*32+d]; vt bf16 [n, h, d, keypad]; out bf16.
// f32 scores in LDS for exact softmax; bf16 probability tile for P@V.
// ---------------------------------------------------------------------------
__global__ void __launch_bounds__(32)
k_attn(const __bf16* __restrict__ qh, const __bf16* __restrict__ kh,
       const __bf16* __restrict__ vt, __bf16* __restrict__ out,
       int Lq, int Lk, int KPAD)
{
  __shared__ float  S [16 * KMAX];
  __shared__ __bf16 Sb[16 * KMAX];
  const int lane = threadIdx.x;
  const int lr   = lane & 15;
  const int half = lane >> 4;
  const int qt = blockIdx.x, h = blockIdx.y, n = blockIdx.z;
  const int q0 = qt * 16;

  // Q fragment (clamped row; padded rows discarded at store)
  const int qrow = (q0 + lr < Lq) ? (q0 + lr) : (Lq - 1);
  const v16bf aq = load_abf(qh + (((size_t)n * Lq + qrow) * HEADS + h) * HD, half);

  // ---- scores S[16][KPAD] = (Q Kt)/16
  for (int k0 = 0; k0 < KPAD; k0 += 16) {
    const int key = k0 + lr;
    v16bf bk;
    if (key < Lk) {
      bk = *(const v16bf*)(kh + (((size_t)n * Lk + key) * HEADS + h) * HD + (half << 4));
    } else {
      const __bf16 z = __builtin_bit_cast(__bf16, (unsigned short)0);
#pragma unroll
      for (int e = 0; e < 16; ++e) bk[e] = z;
    }
    v8f s = {};
    s = WMMA_BF16(aq, bk, s);
#pragma unroll
    for (int r = 0; r < 8; ++r)
      S[(r + half * 8) * KPAD + k0 + lr] = s[r] * INV_SCALE;
  }
  __syncthreads();

  // ---- softmax: lane pair (lr, lr+16) splits the columns of row lr;
  //      exponentials staged in f32 LDS (no private arrays -> no scratch),
  //      normalized probabilities written as bf16 into Sb
  {
    float* row = &S[lr * KPAD];
    __bf16* rowb = &Sb[lr * KPAD];
    const int c0 = half * (KPAD >> 1);
    const int c1 = c0 + (KPAD >> 1);
    float m = -3.4e38f;
    for (int c = c0; c < c1; ++c) if (c < Lk) m = fmaxf(m, row[c]);
    m = fmaxf(m, __shfl_xor(m, 16));
    float sum = 0.f;
    for (int c = c0; c < c1; ++c) {
      const float e = (c < Lk) ? __expf(row[c] - m) : 0.f;
      row[c] = e; sum += e;
    }
    sum += __shfl_xor(sum, 16);
    const float inv = 1.f / sum;
    for (int c = c0; c < c1; ++c) rowb[c] = f2bf(row[c] * inv);
  }
  __syncthreads();

  // ---- O = P V : contract 32 keys per WMMA, two N-tiles for head_dim=32
  v8f o0 = {}, o1 = {};
  const __bf16* vr0 = vt + (((size_t)n * HEADS + h) * HD + lr)      * KPAD;
  const __bf16* vr1 = vt + (((size_t)n * HEADS + h) * HD + 16 + lr) * KPAD;
  for (int k0 = 0; k0 < KPAD; k0 += 32) {
    const v16bf ap = load_abf(&Sb[lr * KPAD + k0], half);
    const v16bf b0 = *(const v16bf*)(vr0 + k0 + (half << 4));
    const v16bf b1 = *(const v16bf*)(vr1 + k0 + (half << 4));
    o0 = WMMA_BF16(ap, b0, o0);
    o1 = WMMA_BF16(ap, b1, o1);
  }
#pragma unroll
  for (int r = 0; r < 8; ++r) {
    const int row = q0 + r + half * 8;
    if (row < Lq) {
      const size_t base = ((size_t)n * Lq + row) * EMBED + h * HD;
      out[base + lr]      = f2bf(o0[r]);
      out[base + 16 + lr] = f2bf(o1[r]);
    }
  }
}

// ---------------------------------------------------------------------------
// LayerNorm(a + b) * g + beta -> f32 copy (residual chain) + bf16 copy (GEMM A)
// ---------------------------------------------------------------------------
__global__ void __launch_bounds__(256)
k_add_ln(const float* __restrict__ a, const float* __restrict__ b,
         const float* __restrict__ g, const float* __restrict__ beta,
         float* __restrict__ out_f, __bf16* __restrict__ out_b)
{
  __shared__ float red[256];
  const int r = blockIdx.x, t = threadIdx.x;
  const size_t idx = (size_t)r * EMBED + t;
  const float v = a[idx] + b[idx];
  red[t] = v; __syncthreads();
  for (int s = 128; s > 0; s >>= 1) { if (t < s) red[t] += red[t + s]; __syncthreads(); }
  const float mean = red[0] * (1.f / EMBED);
  __syncthreads();
  const float d = v - mean;
  red[t] = d * d; __syncthreads();
  for (int s = 128; s > 0; s >>= 1) { if (t < s) red[t] += red[t + s]; __syncthreads(); }
  const float var = red[0] * (1.f / EMBED);
  const float y = d * rsqrtf(var + LN_EPSF) * g[t] + beta[t];
  if (out_f) out_f[idx] = y;
  if (out_b) out_b[idx] = f2bf(y);
}

// ---------------------------------------------------------------------------
// First MLP-encoder layer (fused obs gather + 0.1 factor), bf16 output.
// mode 0: item (R=384, din=3)   mode 1: placements (R=19200, din=6)
// ---------------------------------------------------------------------------
__global__ void __launch_bounds__(256)
k_enc1(const float* __restrict__ obs, const float* __restrict__ w,
       const float* __restrict__ b, __bf16* __restrict__ out, int mode)
{
  const int idx = blockIdx.x * blockDim.x + threadIdx.x;
  const int R = mode ? (BATCH * L_PL) : (BATCH * L_ITEM);
  if (idx >= R * 32) return;
  const int r = idx >> 5, j = idx & 31;
  const float* src;
  int din;
  if (mode) {
    const int nn = r / L_PL, q = r % L_PL, bin = q / 100, p = q % 100;
    src = obs + (size_t)nn * OBS_DIM + bin * SINGLE + AREA + 18 + p * 6;
    din = 6;
  } else {
    const int nn = r / L_ITEM, t = r % L_ITEM;
    src = obs + (size_t)nn * OBS_DIM + AREA + t * 3;
    din = 3;
  }
  float acc = b[j];
  for (int k = 0; k < din; ++k) acc += src[k] * FACTORF * w[k * 32 + j];
  out[(size_t)r * 32 + j] = f2bf((acc > 0.f) ? acc : 0.01f * acc);
}

// ===========================================================================
// Host-side orchestration
// ===========================================================================
namespace {

struct BlkP {
  const float *fco_b, *ff1_b, *ff2_b;
  const float *ln1_b, *ln1_g, *ln2_b, *ln2_g;
  const __bf16 *fco_w, *ff1_w, *ff2_w, *wk, *wq, *wv;  // packed
};

struct Scratch {
  __bf16 *qh, *kh, *vh, *vt, *att, *ffh, *x1b;
  float  *fco, *x1;
};

inline const float* F(void* const* d_in, int i) { return (const float*)d_in[i]; }

inline void gemm(hipStream_t s, const __bf16* A, const __bf16* Wp, const float* bias,
                 float* Cf, __bf16* Cb, int M, int N, int K, int act) {
  if ((N & 63) == 0)
    k_gemm<4><<<dim3(N / 64, M / 32), 32, 0, s>>>(A, Wp, bias, Cf, Cb, M, N, K, act);
  else
    k_gemm<2><<<dim3(N / 32, M / 32), 32, 0, s>>>(A, Wp, bias, Cf, Cb, M, N, K, act);
}

// one transformer block: out = TB(q, kv)
// q_f/q_b: f32 + bf16 copies of the query stream; kv_b: bf16 key/value stream
inline void tblock(hipStream_t s, const float* q_f, const __bf16* q_b,
                   const __bf16* kv_b, const BlkP& p,
                   float* out_f, __bf16* out_b, int Lq, int Lk, const Scratch& w) {
  const int Rq = BATCH * Lq, Rk = BATCH * Lk;
  gemm(s, q_b,  p.wq, nullptr, nullptr, w.qh, Rq * HEADS, HD, HD, 0);
  gemm(s, kv_b, p.wk, nullptr, nullptr, w.kh, Rk * HEADS, HD, HD, 0);
  gemm(s, kv_b, p.wv, nullptr, nullptr, w.vh, Rk * HEADS, HD, HD, 0);

  const int qtiles = (Lq + 15) / 16;
  const int kpad   = ((Lk + 31) / 32) * 32;
  const int vtot = BATCH * HEADS * HD * kpad;
  k_vtrans<<<(vtot + 255) / 256, 256, 0, s>>>(w.vh, w.vt, Lk, kpad);
  k_attn<<<dim3(qtiles, HEADS, BATCH), 32, 0, s>>>(w.qh, w.kh, w.vt, w.att, Lq, Lk, kpad);

  gemm(s, w.att, p.fco_w, p.fco_b, w.fco, nullptr, Rq, EMBED, EMBED, 0);
  k_add_ln<<<Rq, 256, 0, s>>>(w.fco, q_f, p.ln1_g, p.ln1_b, w.x1, w.x1b);
  gemm(s, w.x1b, p.ff1_w, p.ff1_b, nullptr, w.ffh, Rq, 4 * EMBED, EMBED, 1);
  gemm(s, w.ffh, p.ff2_w, p.ff2_b, w.fco, nullptr, Rq, EMBED, 4 * EMBED, 0);  // reuse fco
  k_add_ln<<<Rq, 256, 0, s>>>(w.fco, w.x1, p.ln2_g, p.ln2_b, out_f, out_b);
}

} // namespace

extern "C" void kernel_launch(void* const* d_in, const int* in_sizes, int n_in,
                              void* d_out, int out_size, void* d_ws, size_t ws_size,
                              hipStream_t stream) {
  (void)in_sizes; (void)n_in; (void)out_size; (void)ws_size;

  const float* obs = F(d_in, 0);
  const float* ie0_b = F(d_in, 1); const float* ie0_w = F(d_in, 2);
  const float* ie1_b = F(d_in, 3); const float* ie1_w = F(d_in, 4);
  const int LBASE = 5;
  const int PBASE = LBASE + LAYERS * 4 * 13;
  const float* pe0_b = F(d_in, PBASE + 0); const float* pe0_w = F(d_in, PBASE + 1);
  const float* pe1_b = F(d_in, PBASE + 2); const float* pe1_w = F(d_in, PBASE + 3);

  // ---- workspace carve-up (bytes, 256B aligned chunks) ----
  char* base = (char*)d_ws;
  size_t off = 0;
  auto takeb = [&](size_t bytes) {
    char* p = base + off;
    off += (bytes + 255) & ~(size_t)255;
    return p;
  };
  const size_t N_ITEM = (size_t)BATCH * L_ITEM * EMBED;   //    98,304
  const size_t N_PL   = (size_t)BATCH * L_PL   * EMBED;   // 4,915,200

  __bf16* h32i    = (__bf16*)takeb((size_t)BATCH * L_ITEM * 32 * 2);
  __bf16* h32p    = (__bf16*)takeb((size_t)BATCH * L_PL * 32 * 2);
  float*  item_a  = (float*) takeb(N_ITEM * 4);
  float*  item_b  = (float*) takeb(N_ITEM * 4);
  float*  pl_a    = (float*) takeb(N_PL * 4);
  float*  pl_b    = (float*) takeb(N_PL * 4);
  __bf16* item_ab = (__bf16*)takeb(N_ITEM * 2);
  __bf16* item_bb = (__bf16*)takeb(N_ITEM * 2);
  __bf16* pl_ab   = (__bf16*)takeb(N_PL * 2);
  __bf16* pl_bb   = (__bf16*)takeb(N_PL * 2);
  Scratch sc;
  sc.qh  = (__bf16*)takeb(N_PL * 2);
  sc.kh  = (__bf16*)takeb(N_PL * 2);
  sc.vh  = (__bf16*)takeb(N_PL * 2);
  sc.vt  = (__bf16*)takeb((size_t)BATCH * HEADS * HD * KMAX * 2);
  sc.att = (__bf16*)takeb(N_PL * 2);
  sc.ffh = (__bf16*)takeb((size_t)BATCH * L_PL * 4 * EMBED * 2);
  sc.x1b = (__bf16*)takeb(N_PL * 2);
  sc.fco = (float*) takeb(N_PL * 4);
  sc.x1  = (float*) takeb(N_PL * 4);

  // ---- pack all weights to bf16 fragment layout ----
  auto pack = [&](const float* W, int K, int N) -> const __bf16* {
    __bf16* dst = (__bf16*)takeb((size_t)K * N * 2);
    k_pack_w<<<(K * N + 255) / 256, 256, 0, stream>>>(W, dst, K, N);
    return dst;
  };
  const __bf16* ie1_wp = pack(ie1_w, 32, EMBED);
  const __bf16* pe1_wp = pack(pe1_w, 32, EMBED);

  BlkP blks[LAYERS][4];
  for (int l = 0; l < LAYERS; ++l) {
    for (int j = 0; j < 4; ++j) {
      const int b = LBASE + (l * 4 + j) * 13;
      BlkP& p = blks[l][j];
      p.fco_b = F(d_in, b + 0);
      p.fco_w = pack(F(d_in, b + 1), EMBED, EMBED);
      p.ff1_b = F(d_in, b + 2);
      p.ff1_w = pack(F(d_in, b + 3), EMBED, 4 * EMBED);
      p.ff2_b = F(d_in, b + 4);
      p.ff2_w = pack(F(d_in, b + 5), 4 * EMBED, EMBED);
      p.ln1_b = F(d_in, b + 6);  p.ln1_g = F(d_in, b + 7);
      p.ln2_b = F(d_in, b + 8);  p.ln2_g = F(d_in, b + 9);
      p.wk = pack(F(d_in, b + 10), HD, HD);
      p.wq = pack(F(d_in, b + 11), HD, HD);
      p.wv = pack(F(d_in, b + 12), HD, HD);
    }
  }

  // ---- input encoders ----
  k_enc1<<<(BATCH * L_ITEM * 32 + 255) / 256, 256, 0, stream>>>(obs, ie0_w, ie0_b, h32i, 0);
  gemm(stream, h32i, ie1_wp, ie1_b, item_a, item_ab, BATCH * L_ITEM, EMBED, 32, 0);
  k_enc1<<<(BATCH * L_PL * 32 + 255) / 256, 256, 0, stream>>>(obs, pe0_w, pe0_b, h32p, 1);
  gemm(stream, h32p, pe1_wp, pe1_b, pl_a, pl_ab, BATCH * L_PL, EMBED, 32, 0);

  // ---- encoder stack ----
  for (int l = 0; l < LAYERS; ++l) {
    const BlkP& p_ems_on_item = blks[l][0];
    const BlkP& p_ems_self    = blks[l][1];
    const BlkP& p_item_on_ems = blks[l][2];
    const BlkP& p_item_self   = blks[l][3];

    // item_e = TB(item, item);  ems_e = TB(pl, pl)
    tblock(stream, item_a, item_ab, item_ab, p_item_self, item_b, item_bb, L_ITEM, L_ITEM, sc);
    tblock(stream, pl_a,   pl_ab,   pl_ab,   p_ems_self,  pl_b,   pl_bb,   L_PL,   L_PL,   sc);
    // ems_on_item = TB(q=ems_e, kv=item_e);  item_on_ems = TB(q=item_e, kv=ems_e)
    tblock(stream, pl_b,   pl_bb,   item_bb, p_ems_on_item, pl_a,   pl_ab,   L_PL,   L_ITEM, sc);
    tblock(stream, item_b, item_bb, pl_bb,   p_item_on_ems, item_a, item_ab, L_ITEM, L_PL,   sc);
  }

  // ---- outputs: (item_embedding, placement_embedding) concatenated ----
  hipMemcpyAsync(d_out, item_a, N_ITEM * sizeof(float),
                 hipMemcpyDeviceToDevice, stream);
  hipMemcpyAsync((float*)d_out + N_ITEM, pl_a, N_PL * sizeof(float),
                 hipMemcpyDeviceToDevice, stream);
}
